// LocalDownsampleFlexAttnProcessor_31576599560565
// MI455X (gfx1250) — compile-verified, matrix operand order checked by reading, not run on hardware
//
#include <hip/hip_runtime.h>

// ---------------- CDNA5 (gfx1250, wave32) types ----------------
typedef __attribute__((ext_vector_type(16))) __bf16 v16bf;
typedef __attribute__((ext_vector_type(8)))  float  v8f;

union Frag { v16bf v; int4 q[2]; };

// ---------------- problem constants (from setup_inputs) ----------------
#define DIMC  3072
#define SEQ   1536
#define HD    128
#define TXTC  512
#define KTOT  1600        // 1536 + 64 downsampled keys
#define NHEAD 24

static __device__ __forceinline__ __bf16 f2bf(float x) {
    unsigned u = __builtin_bit_cast(unsigned, x);
    unsigned r = u + 0x7FFFu + ((u >> 16) & 1u);   // round-to-nearest-even
    unsigned short h = (unsigned short)(r >> 16);
    return __builtin_bit_cast(__bf16, h);
}
static __device__ __forceinline__ float bf2f(__bf16 b) {
    unsigned short h = __builtin_bit_cast(unsigned short, b);
    return __builtin_bit_cast(float, (unsigned)h << 16);
}

// ---------------- 1) weight f32 -> bf16 with transpose: Wt[n][k] ----------------
__global__ __launch_bounds__(256) void wt_conv(const float* __restrict__ Wq,
                                               const float* __restrict__ Wk,
                                               const float* __restrict__ Wv,
                                               const float* __restrict__ Wo,
                                               __bf16* __restrict__ out) {
    __shared__ float t[32][33];
    const float* W = (blockIdx.z == 0) ? Wq : (blockIdx.z == 1) ? Wk
                   : (blockIdx.z == 2) ? Wv : Wo;
    __bf16* o = out + (size_t)blockIdx.z * DIMC * DIMC;
    const int tx = threadIdx.x, ty = threadIdx.y;      // 32 x 8
    const int n0 = blockIdx.x * 32, k0 = blockIdx.y * 32;
#pragma unroll
    for (int i = 0; i < 4; ++i)
        t[ty + i * 8][tx] = W[(size_t)(k0 + ty + i * 8) * DIMC + n0 + tx];
    __syncthreads();
#pragma unroll
    for (int i = 0; i < 4; ++i)
        o[(size_t)(n0 + ty + i * 8) * DIMC + k0 + tx] = f2bf(t[tx][ty + i * 8]);
}

// ---------------- 2) X f32 -> bf16 ----------------
__global__ __launch_bounds__(256) void xconv(const float* __restrict__ X, __bf16* __restrict__ Xb) {
    int i = blockIdx.x * 256 + threadIdx.x;
    Xb[i] = f2bf(X[i]);
}

// ---------------- 3) WMMA GEMM: C = A(MxK,k-contig) * Bt(NxK,k-contig)^T + bias --------
// Double-buffered LDS pipeline, 64-deep K tiles, 8 waves x (64x32) per 128x128 block.
// MODE 0: bf16 out, per-head row-major:  Out[h*hs + m*128 + d]   (Q and K)
// MODE 1: bf16 out, per-head col-major:  Out[h*hs + d*KTOT + m]  (V transposed)
// MODE 2: f32  out, row-major M x N:     Out[m*DIMC + n]         (final projection)
#define PITCH 72   // 64 + 8 pad: 144B rows spread across the 64 LDS banks
template <int MODE>
__global__ __launch_bounds__(256) void gemm_bf16(const __bf16* __restrict__ A,
                                                 const __bf16* __restrict__ Bt,
                                                 const float* __restrict__ bias,
                                                 void* __restrict__ Out, int hs) {
    __shared__ __align__(16) __bf16 As[2][128 * PITCH];   // 2 x 18KB
    __shared__ __align__(16) __bf16 Bs[2][128 * PITCH];
    const int tid  = threadIdx.x;
    const int lane = tid & 31, wave = tid >> 5;
    const int mBlock = blockIdx.y * 128, nBlock = blockIdx.x * 128;
    const int mw = (wave >> 2) * 64, nw = (wave & 3) * 32;   // wave tile: 64 x 32
    const int lrow = tid >> 1;                 // 0..127 (tile row loaded by this thread)
    const int lq   = (tid & 1) * 32;           // element offset within 64-deep K slab
    const int half = (lane >> 4) * 8;          // wave32 A/B fragment K offset
    const int l15  = lane & 15;

    const __bf16* gA = A  + (size_t)(mBlock + lrow) * DIMC + lq;
    const __bf16* gB = Bt + (size_t)(nBlock + lrow) * DIMC + lq;

    // ---- preload tile 0 ----
    int4 pa[4], pb[4];
#pragma unroll
    for (int u = 0; u < 4; ++u) {
        pa[u] = *(const int4*)(gA + u * 8);
        pb[u] = *(const int4*)(gB + u * 8);
    }
#pragma unroll
    for (int u = 0; u < 4; ++u) {
        *(int4*)(As[0] + lrow * PITCH + lq + u * 8) = pa[u];
        *(int4*)(Bs[0] + lrow * PITCH + lq + u * 8) = pb[u];
    }
    __syncthreads();

    v8f acc[4][2] = {};
    int buf = 0;

    for (int kt = 0; kt < DIMC / 64; ++kt) {           // 48 iterations
        // L2 -> near-cache warm-up for the stream 8 tiles ahead
        if (kt + 8 < DIMC / 64) {
            __builtin_prefetch(gA + (kt + 8) * 64, 0, 1);
            __builtin_prefetch(gB + (kt + 8) * 64, 0, 1);
        }
        // issue next tile's global loads (overlap with WMMA below)
        if (kt + 1 < DIMC / 64) {
            const __bf16* na = gA + (kt + 1) * 64;
            const __bf16* nb = gB + (kt + 1) * 64;
#pragma unroll
            for (int u = 0; u < 4; ++u) {
                pa[u] = *(const int4*)(na + u * 8);
                pb[u] = *(const int4*)(nb + u * 8);
            }
        }
        // compute current tile: 2 K-chunks x 8 WMMA
#pragma unroll
        for (int c = 0; c < 2; ++c) {
            Frag af[4], bg[2];
#pragma unroll
            for (int i = 0; i < 4; ++i) {
                const __bf16* p = As[buf] + (mw + i * 16 + l15) * PITCH + c * 32 + half;
                af[i].q[0] = *(const int4*)p;  af[i].q[1] = *(const int4*)(p + 16);
            }
#pragma unroll
            for (int j = 0; j < 2; ++j) {
                const __bf16* p = Bs[buf] + (nw + j * 16 + l15) * PITCH + c * 32 + half;
                bg[j].q[0] = *(const int4*)p;  bg[j].q[1] = *(const int4*)(p + 16);
            }
#pragma unroll
            for (int i = 0; i < 4; ++i)
#pragma unroll
                for (int j = 0; j < 2; ++j)
                    acc[i][j] = __builtin_amdgcn_wmma_f32_16x16x32_bf16(
                        false, af[i].v, false, bg[j].v, (short)0, acc[i][j], false, false);
        }
        // commit prefetch into the other buffer, single barrier per iteration
        if (kt + 1 < DIMC / 64) {
            int nb_ = buf ^ 1;
#pragma unroll
            for (int u = 0; u < 4; ++u) {
                *(int4*)(As[nb_] + lrow * PITCH + lq + u * 8) = pa[u];
                *(int4*)(Bs[nb_] + lrow * PITCH + lq + u * 8) = pb[u];
            }
            __syncthreads();
            buf = nb_;
        }
    }

#pragma unroll
    for (int i = 0; i < 4; ++i)
#pragma unroll
        for (int j = 0; j < 2; ++j)
#pragma unroll
            for (int r = 0; r < 8; ++r) {
                int m = mBlock + mw + i * 16 + r + ((lane >> 4) << 3);
                int n = nBlock + nw + j * 16 + l15;
                float val = acc[i][j][r] + bias[n];
                if constexpr (MODE == 0) {
                    int h = n >> 7, d = n & 127;
                    ((__bf16*)Out)[(size_t)h * hs + (size_t)m * HD + d] = f2bf(val);
                } else if constexpr (MODE == 1) {
                    int h = n >> 7, d = n & 127;
                    ((__bf16*)Out)[(size_t)h * hs + (size_t)d * KTOT + m] = f2bf(val);
                } else {
                    ((float*)Out)[(size_t)m * DIMC + n] = val;
                }
            }
}

// ---------------- 4) spatial-weight downsample of K / V image keys ----------------
__global__ __launch_bounds__(128) void downsample_kv(const float* __restrict__ sw,
                                                     __bf16* __restrict__ Kall,
                                                     __bf16* __restrict__ Vt) {
    const int d = threadIdx.x;            // 0..127
    const int p = blockIdx.x;             // 0..63 coarse pixel
    const int h = blockIdx.y;
    const int oy = p >> 3, ox = p & 7;
    float acc = 0.f;
    if (blockIdx.z == 0) {
        const __bf16* base = Kall + (size_t)h * (KTOT * HD);
#pragma unroll
        for (int fy = 0; fy < 4; ++fy)
#pragma unroll
            for (int fx = 0; fx < 4; ++fx) {
                int pix = (oy * 4 + fy) * 32 + ox * 4 + fx;
                acc += bf2f(base[(size_t)(TXTC + pix) * HD + d]) * sw[fy * 4 + fx];
            }
        Kall[(size_t)h * (KTOT * HD) + (size_t)(SEQ + p) * HD + d] = f2bf(acc);
    } else {
        const __bf16* base = Vt + (size_t)h * (HD * KTOT) + (size_t)d * KTOT;
#pragma unroll
        for (int fy = 0; fy < 4; ++fy)
#pragma unroll
            for (int fx = 0; fx < 4; ++fx) {
                int pix = (oy * 4 + fy) * 32 + ox * 4 + fx;
                acc += bf2f(base[TXTC + pix]) * sw[fy * 4 + fx];
            }
        Vt[(size_t)h * (HD * KTOT) + (size_t)d * KTOT + SEQ + p] = f2bf(acc);
    }
}

// ---------------- 5) flash attention: per (head, 64-row q block), 4 waves ----------------
__global__ __launch_bounds__(128) void attn_kernel(const __bf16* __restrict__ Qh,
                                                   const __bf16* __restrict__ Kall,
                                                   const __bf16* __restrict__ Vt,
                                                   __bf16* __restrict__ A2) {
    __shared__ __align__(16) __bf16 Pbuf[4][16 * 40];   // per-wave P staging (C->A relayout)
    const int lane = threadIdx.x & 31, wave = threadIdx.x >> 5;
    const int h  = blockIdx.y;
    const int q0 = blockIdx.x * 64 + wave * 16;
    const int half = (lane >> 4) * 8;
    const int l15  = lane & 15;
    const float scale = 0.08838834764831845f;   // 1/sqrt(128)

    // resident Q fragments (16 rows x 128d = 4 K-chunks)
    const __bf16* Qbase = Qh + (size_t)h * (SEQ * HD) + (size_t)q0 * HD;
    Frag qf[4];
#pragma unroll
    for (int c = 0; c < 4; ++c) {
        const __bf16* p = Qbase + l15 * HD + c * 32 + half;
        qf[c].q[0] = *(const int4*)p;  qf[c].q[1] = *(const int4*)(p + 16);
    }

    float rmax[8], rsum[8];
    v8f o[8] = {};
#pragma unroll
    for (int r = 0; r < 8; ++r) { rmax[r] = -1e30f; rsum[r] = 0.f; }

    const __bf16* Kb = Kall + (size_t)h * (KTOT * HD);
    const __bf16* Vb = Vt   + (size_t)h * (HD * KTOT);
    __bf16* Pw = Pbuf[wave];

    for (int kb = 0; kb < KTOT; kb += 32) {
        // ---- S = Q K^T for 32 keys (two 16x16 tiles) ----
        v8f s0 = {}, s1 = {};
#pragma unroll
        for (int c = 0; c < 4; ++c) {
            Frag kf0, kf1;
            const __bf16* p0 = Kb + (size_t)(kb + l15) * HD + c * 32 + half;
            const __bf16* p1 = p0 + 16 * HD;
            kf0.q[0] = *(const int4*)p0;  kf0.q[1] = *(const int4*)(p0 + 16);
            kf1.q[0] = *(const int4*)p1;  kf1.q[1] = *(const int4*)(p1 + 16);
            s0 = __builtin_amdgcn_wmma_f32_16x16x32_bf16(false, qf[c].v, false, kf0.v, (short)0, s0, false, false);
            s1 = __builtin_amdgcn_wmma_f32_16x16x32_bf16(false, qf[c].v, false, kf1.v, (short)0, s1, false, false);
        }
        // ---- online softmax (row stats shared across each 16-lane half-wave) ----
#pragma unroll
        for (int r = 0; r < 8; ++r) {
            float a = s0[r] * scale, b = s1[r] * scale;
            float mx = fmaxf(a, b);
#pragma unroll
            for (int off = 1; off < 16; off <<= 1) mx = fmaxf(mx, __shfl_xor(mx, off, 32));
            float nm   = fmaxf(rmax[r], mx);
            float corr = __expf(rmax[r] - nm);
            rmax[r] = nm;
            float p0 = __expf(a - nm), p1 = __expf(b - nm);
            float ps = p0 + p1;
#pragma unroll
            for (int off = 1; off < 16; off <<= 1) ps += __shfl_xor(ps, off, 32);
            rsum[r] = rsum[r] * corr + ps;
#pragma unroll
            for (int t = 0; t < 8; ++t) o[t][r] *= corr;
            int m = r + ((lane >> 4) << 3);
            Pw[m * 40 + l15]      = f2bf(p0);
            Pw[m * 40 + 16 + l15] = f2bf(p1);
        }
        // ---- reload P as an A-fragment (DS in-order within wave) ----
        Frag pf;
        {
            const __bf16* p = Pw + l15 * 40 + half;
            pf.q[0] = *(const int4*)p;  pf.q[1] = *(const int4*)(p + 16);
        }
        // ---- O += P V  (V stored transposed: key-contiguous) ----
#pragma unroll
        for (int t = 0; t < 8; ++t) {
            Frag vf;
            const __bf16* p = Vb + (size_t)(t * 16 + l15) * KTOT + kb + half;
            vf.q[0] = *(const int4*)p;  vf.q[1] = *(const int4*)(p + 16);
            o[t] = __builtin_amdgcn_wmma_f32_16x16x32_bf16(false, pf.v, false, vf.v, (short)0, o[t], false, false);
        }
    }

    // ---- normalize and emit bf16 attention output, row-major [S][DIM] ----
#pragma unroll
    for (int r = 0; r < 8; ++r) {
        float inv = 1.0f / rsum[r];
        int m = q0 + r + ((lane >> 4) << 3);
#pragma unroll
        for (int t = 0; t < 8; ++t) {
            int d = t * 16 + l15;
            A2[(size_t)m * DIMC + h * HD + d] = f2bf(o[t][r] * inv);
        }
    }
}

// ---------------- host-side orchestration ----------------
extern "C" void kernel_launch(void* const* d_in, const int* in_sizes, int n_in,
                              void* d_out, int out_size, void* d_ws, size_t ws_size,
                              hipStream_t stream) {
    const float* X  = (const float*)d_in[0];
    const float* Wq = (const float*)d_in[1];
    const float* bq = (const float*)d_in[2];
    const float* Wk = (const float*)d_in[3];
    const float* bk = (const float*)d_in[4];
    const float* Wv = (const float*)d_in[5];
    const float* bv = (const float*)d_in[6];
    const float* Wo = (const float*)d_in[7];
    const float* bo = (const float*)d_in[8];
    const float* sw = (const float*)d_in[9];
    (void)in_sizes; (void)n_in; (void)out_size; (void)ws_size;

    char* ws = (char*)d_ws;
    constexpr size_t W_ELEMS = (size_t)DIMC * DIMC;          // per weight matrix
    __bf16* WT   = (__bf16*)ws;                              // 4 * 18,874,368 B
    char*   p    = ws + 4 * W_ELEMS * sizeof(__bf16);
    __bf16* XB   = (__bf16*)p;  p += (size_t)SEQ * DIMC * 2; // 9.4 MB
    __bf16* QH   = (__bf16*)p;  p += (size_t)NHEAD * SEQ * HD * 2;
    __bf16* KALL = (__bf16*)p;  p += (size_t)NHEAD * KTOT * HD * 2;
    __bf16* VT   = (__bf16*)p;  p += (size_t)NHEAD * HD * KTOT * 2;
    __bf16* A2   = (__bf16*)p;  // SEQ * DIMC bf16

    // 1) weights -> bf16, transposed to Wt[n][k]
    wt_conv<<<dim3(DIMC / 32, DIMC / 32, 4), dim3(32, 8), 0, stream>>>(Wq, Wk, Wv, Wo, WT);
    // 2) activations -> bf16
    xconv<<<(SEQ * DIMC) / 256, 256, 0, stream>>>(X, XB);
    // 3) Q/K/V projections (WMMA bf16, f32 accumulate)
    gemm_bf16<0><<<dim3(DIMC / 128, SEQ / 128), 256, 0, stream>>>(XB, WT + 0 * W_ELEMS, bq, QH,   SEQ * HD);
    gemm_bf16<0><<<dim3(DIMC / 128, SEQ / 128), 256, 0, stream>>>(XB, WT + 1 * W_ELEMS, bk, KALL, KTOT * HD);
    gemm_bf16<1><<<dim3(DIMC / 128, SEQ / 128), 256, 0, stream>>>(XB, WT + 2 * W_ELEMS, bv, VT,   HD * KTOT);
    // 4) downsampled keys/values (rows 1536..1599)
    downsample_kv<<<dim3(64, NHEAD, 2), 128, 0, stream>>>(sw, KALL, VT);
    // 5) flash attention
    attn_kernel<<<dim3(SEQ / 64, NHEAD), 128, 0, stream>>>(QH, KALL, VT, A2);
    // 6) output projection -> f32 d_out
    gemm_bf16<2><<<dim3(DIMC / 128, SEQ / 128), 256, 0, stream>>>(A2, WT + 3 * W_ELEMS, bo, d_out, 0);
}
